// Transformer_47674136985732
// MI455X (gfx1250) — matrix operand order repaired and checked by reading, not compiled, not run on hardware
//
#include <hip/hip_runtime.h>
#include <hip/hip_bf16.h>

// ---------------------------------------------------------------------------
// MI455X (gfx1250) decoder-only transformer forward.
// - Encoder is dead code w.r.t. the reference output (cross-attn uses kv=dec),
//   so it is skipped entirely.
// - All matmuls: v_wmma_f32_16x16x32_bf16 (wave32, f32 accumulate).
// - GEMM mainloop is double-buffered with global_load_async_to_lds_b128
//   (ASYNCcnt-tracked, VGPR-bypassing) -- builtin prototype is
//   (v4i addrspace(1)*, v4i addrspace(3)*, imm offset, imm cpol).
// ---------------------------------------------------------------------------

typedef __attribute__((ext_vector_type(16))) __bf16       v16bf;
typedef __attribute__((ext_vector_type(8)))  float        v8f;
typedef __attribute__((ext_vector_type(4)))  unsigned int u32x4;
typedef __attribute__((ext_vector_type(4)))  float        f32x4;
typedef __attribute__((ext_vector_type(4)))  int          v4i_t;

union FragBF { u32x4 q[2]; v16bf v; };

#define WMMA_BF16(a, b, c) \
  __builtin_amdgcn_wmma_f32_16x16x32_bf16(false, (a), false, (b), (short)0, (c), false, false)

#if __has_builtin(__builtin_amdgcn_global_load_async_to_lds_b128) && \
    __has_builtin(__builtin_amdgcn_s_wait_asynccnt)
#define HAVE_ASYNC_LDS 1
typedef __attribute__((address_space(1))) v4i_t* gv4i_p;  // global int4*
typedef __attribute__((address_space(3))) v4i_t* lv4i_p;  // LDS int4*
#else
#define HAVE_ASYNC_LDS 0
#endif

// Copy 16B global -> LDS. Async (VGPR-bypassing, ASYNCcnt-tracked) when
// available; falls back to load+ds_store otherwise. Per-lane addresses.
__device__ inline void copy16_to_lds(const __bf16* g, __bf16* l) {
#if HAVE_ASYNC_LDS
  // Integer round-trip = the exact lowering of addrspacecast on AMDGPU:
  // AS(1) keeps the 64-bit value; AS(3) is the low-32 LDS byte offset.
  gv4i_p gp = (gv4i_p)(unsigned long long)(uintptr_t)g;
  lv4i_p lp = (lv4i_p)(unsigned int)(uintptr_t)l;
  __builtin_amdgcn_global_load_async_to_lds_b128(gp, lp, 0, 0);
#else
  *(u32x4*)l = *(const u32x4*)g;
#endif
}

__device__ inline void async_fence() {
#if HAVE_ASYNC_LDS
  __builtin_amdgcn_s_wait_asynccnt(0);
#endif
}

__device__ inline v8f vzero8() {
  v8f z;
#pragma unroll
  for (int r = 0; r < 8; ++r) z[r] = 0.0f;
  return z;
}

// A-matrix fragment (16x32 bf16), CDNA5 ISA 7.12.2 layout:
// lane<16 : row=lane,    K = {0..7, 16..23}; lane>=16: row=lane-16, K={8..15,24..31}
// Two contiguous 16B reads. Source layout [row][k], row stride ld (16B-multiple).
__device__ inline v16bf load_frag_a(const __bf16* p0, int ld) {
  const int lane = threadIdx.x & 31;
  const int r    = lane & 15;
  const int kb   = (lane >> 4) << 3;
  const __bf16* p = p0 + r * ld + kb;
  FragBF f;
  f.q[0] = *(const u32x4*)(p);
  f.q[1] = *(const u32x4*)(p + 16);
  return f.v;
}

// B-matrix fragment (32x16 bf16): lane<16: col=lane, K={0..7,16..23}; lane>=16:
// col=lane-16, K={8..15,24..31}. Two 16B reads. Source layout [n][k].
__device__ inline v16bf load_frag_b(const __bf16* p0, int ld) {
  const int lane = threadIdx.x & 31;
  const int n    = lane & 15;
  const int ko   = (lane >> 4) << 3;
  const __bf16* p = p0 + n * ld + ko;
  FragBF f;
  f.q[0] = *(const u32x4*)(p);
  f.q[1] = *(const u32x4*)(p + 16);
  return f.v;
}

// ---------------------------------------------------------------------------
// Positional encoding + cast: xF = dec_x + PE, xB = bf16(xF)
// ---------------------------------------------------------------------------
__global__ void k_add_pe(const float* __restrict__ x, float* __restrict__ xF,
                         __bf16* __restrict__ xB, int total) {
  int idx = blockIdx.x * blockDim.x + threadIdx.x;
  if (idx >= total) return;
  int l = (idx >> 9) & 511;
  int d = idx & 511;
  int i = d >> 1;
  float freq = __expf((float)i * (9.21034037f / 512.0f));  // 10000^(i/512)
  float arg  = (float)l / freq;
  float pe   = (d & 1) ? __cosf(arg) : __sinf(arg);
  float v = x[idx] + pe;
  xF[idx] = v;
  xB[idx] = (__bf16)v;
}

// ---------------------------------------------------------------------------
// GEMM: C[M,N] = act(A_bf16[M,K] @ W_f32[K,N] + bias) (+residual).
// 256 threads = 8 waves (2x4); block tile 128x128x32; wave tile 64x32 (4x2 WMMA).
// Double-buffered LDS; A tile staged with async global->LDS copies; W tile
// loaded early, converted f32->bf16 + transposed-stored after the WMMA block.
// Epilogue fully specialized via template flags (no per-element branches).
// ---------------------------------------------------------------------------
__device__ inline void store_bs(__bf16* Bsb, int bc, int bk,
                                f32x4 w0, f32x4 w1, f32x4 w2, f32x4 w3) {
  f32x4 w[4] = {w0, w1, w2, w3};
#pragma unroll
  for (int v = 0; v < 4; ++v) {
    Bsb[(bc + v * 4 + 0) * 48 + bk] = (__bf16)w[v].x;
    Bsb[(bc + v * 4 + 1) * 48 + bk] = (__bf16)w[v].y;
    Bsb[(bc + v * 4 + 2) * 48 + bk] = (__bf16)w[v].z;
    Bsb[(bc + v * 4 + 3) * 48 + bk] = (__bf16)w[v].w;
  }
}

template <bool BIAS, bool RELU, bool RES, bool OUTF, bool OUTB>
__global__ __launch_bounds__(256) void k_gemm(
    const __bf16* __restrict__ A, const float* __restrict__ W,
    const float* __restrict__ bias, const float* __restrict__ residual,
    float* __restrict__ outF, __bf16* __restrict__ outB,
    int M, int N, int K) {
  __shared__ __align__(16) __bf16 As[2][128 * 48];  // [m][k], ld=48 (96B rows)
  __shared__ __align__(16) __bf16 Bs[2][128 * 48];  // [n][k] transposed, ld=48
  (void)M;

  const int t    = threadIdx.x;
  const int lane = t & 31, wid = t >> 5;
  const int wm = wid >> 2, wn = wid & 3;
  const int m0 = blockIdx.y * 128, n0 = blockIdx.x * 128;

  const int ar = t >> 1, ac = (t & 1) * 16;  // A staging: 2 threads / row
  const int bk = t >> 3, bc = (t & 7) * 16;  // W staging: 8 threads / row

  const __bf16* gA = A + (size_t)(m0 + ar) * K + ac;   // advance by +32/step
  const float*  gW = W + (size_t)bk * N + n0 + bc;     // advance by +32*N/step

  // ---- prologue: stage k-tile 0 into buffer 0 ----
  copy16_to_lds(gA, &As[0][ar * 48 + ac]);
  copy16_to_lds(gA + 8, &As[0][ar * 48 + ac + 8]);
  {
    f32x4 w0 = *(const f32x4*)(gW + 0);
    f32x4 w1 = *(const f32x4*)(gW + 4);
    f32x4 w2 = *(const f32x4*)(gW + 8);
    f32x4 w3 = *(const f32x4*)(gW + 12);
    store_bs(&Bs[0][0], bc, bk, w0, w1, w2, w3);
  }
  async_fence();
  __syncthreads();

  v8f acc[4][2];
#pragma unroll
  for (int i = 0; i < 4; ++i)
#pragma unroll
    for (int j = 0; j < 2; ++j) acc[i][j] = vzero8();

  const int ksteps = K >> 5;
  for (int ks = 0; ks < ksteps; ++ks) {
    const int cur = ks & 1, nxt = cur ^ 1;
    const bool pending = (ks + 1) < ksteps;

    // Kick off next tile's data movement before the math.
    f32x4 w0, w1, w2, w3;
    if (pending) {
      const __bf16* ga = gA + (size_t)(ks + 1) * 32;
      copy16_to_lds(ga, &As[nxt][ar * 48 + ac]);
      copy16_to_lds(ga + 8, &As[nxt][ar * 48 + ac + 8]);
      const float* gw = gW + (size_t)(ks + 1) * 32 * N;
      w0 = *(const f32x4*)(gw + 0);
      w1 = *(const f32x4*)(gw + 4);
      w2 = *(const f32x4*)(gw + 8);
      w3 = *(const f32x4*)(gw + 12);
      __builtin_prefetch(gw + (size_t)32 * N, 0, 1);  // tile k+2 -> L2
    }

    // WMMA block on current buffer.
    v16bf af[4];
#pragma unroll
    for (int mi = 0; mi < 4; ++mi)
      af[mi] = load_frag_a(&As[cur][(wm * 64 + mi * 16) * 48], 48);
    v16bf bf0 = load_frag_b(&Bs[cur][(wn * 32) * 48], 48);
    v16bf bf1 = load_frag_b(&Bs[cur][(wn * 32 + 16) * 48], 48);
#pragma unroll
    for (int mi = 0; mi < 4; ++mi) {
      acc[mi][0] = WMMA_BF16(af[mi], bf0, acc[mi][0]);
      acc[mi][1] = WMMA_BF16(af[mi], bf1, acc[mi][1]);
    }

    if (pending) store_bs(&Bs[nxt][0], bc, bk, w0, w1, w2, w3);
    async_fence();
    __syncthreads();
  }

  // ---- epilogue (branch-free per instantiation) ----
  const int hi = lane >> 4, nn = lane & 15;
  float bv0 = 0.0f, bv1 = 0.0f;
  if constexpr (BIAS) {
    bv0 = bias[n0 + wn * 32 + nn];
    bv1 = bias[n0 + wn * 32 + 16 + nn];
  }
#pragma unroll
  for (int mi = 0; mi < 4; ++mi)
#pragma unroll
    for (int ni = 0; ni < 2; ++ni)
#pragma unroll
      for (int r = 0; r < 8; ++r) {
        int row = m0 + wm * 64 + mi * 16 + r + 8 * hi;
        int col = n0 + wn * 32 + ni * 16 + nn;
        float v = acc[mi][ni][r];
        if constexpr (BIAS) v += (ni ? bv1 : bv0);
        if constexpr (RELU) v = fmaxf(v, 0.0f);
        size_t off = (size_t)row * N + col;
        if constexpr (RES) v += residual[off];
        if constexpr (OUTF) outF[off] = v;
        if constexpr (OUTB) outB[off] = (__bf16)v;
      }
}

// ---------------------------------------------------------------------------
// Flash attention: one workgroup per (b, h, 128-query tile); 8 waves x 16 rows.
// Online softmax over 32-key blocks; QK^T and P*V through WMMA.
// K block staged via async global->LDS; V staged transposed (needs layout swap).
// ---------------------------------------------------------------------------
__global__ __launch_bounds__(256) void k_attn(
    const __bf16* __restrict__ Q, const __bf16* __restrict__ Kg,
    const __bf16* __restrict__ V, __bf16* __restrict__ O, int causal) {
  __shared__ __align__(16) __bf16 Ks[32 * 72];      // [key][chan], ld=72 (144B)
  __shared__ __align__(16) __bf16 Vt[64 * 48];      // [chan][key], ld=48 (96B)
  __shared__ __align__(16) __bf16 Ps[8 * 16 * 48];  // per-wave P strip [row][key]

  const int t = threadIdx.x, lane = t & 31, wid = t >> 5;
  const int qt = blockIdx.x, h = blockIdx.y, b = blockIdx.z;
  const int qbase = qt * 128 + wid * 16;
  const size_t tok0 = (size_t)b * 512 + qbase;
  const int hi = lane >> 4, nn = lane & 15;

  // Preload Q fragments for this wave's 16 rows (chans 0..63).
  v16bf qf0, qf1;
  {
    const __bf16* qp = Q + tok0 * 512 + (size_t)h * 64;
    qf0 = load_frag_a(qp, 512);
    qf1 = load_frag_a(qp + 32, 512);
  }

  v8f o0 = vzero8(), o1 = vzero8(), o2 = vzero8(), o3 = vzero8();
  float mrow[8], lrow[8];
#pragma unroll
  for (int r = 0; r < 8; ++r) { mrow[r] = -3.0e38f; lrow[r] = 0.0f; }

  const int kr = t >> 3, kc = (t & 7) * 8;  // staging: 8 thr/key, 8 chans each
  __bf16* Pw = Ps + wid * 16 * 48;

  for (int kb = 0; kb < 16; ++kb) {
    __syncthreads();
    {  // stage 32-key block: K async as-is, V transposed through VGPRs
      const size_t ktok = (size_t)b * 512 + kb * 32 + kr;
      const __bf16* gk = Kg + ktok * 512 + (size_t)h * 64 + kc;
      copy16_to_lds(gk, Ks + kr * 72 + kc);
      const __bf16* gv = V + ktok * 512 + (size_t)h * 64 + kc;
      u32x4 v4 = *(const u32x4*)gv;
      const __bf16* vw = (const __bf16*)&v4;
#pragma unroll
      for (int j = 0; j < 8; ++j) Vt[(kc + j) * 48 + kr] = vw[j];
    }
    async_fence();
    __syncthreads();

    // S = Q @ K^T for two 16-key tiles (chans 0..63 = 2 WMMA steps each)
    v8f s0 = vzero8(), s1 = vzero8();
    {
      v16bf kf;
      kf = load_frag_b(Ks + 0 * 72 + 0, 72);   s0 = WMMA_BF16(qf0, kf, s0);
      kf = load_frag_b(Ks + 0 * 72 + 32, 72);  s0 = WMMA_BF16(qf1, kf, s0);
      kf = load_frag_b(Ks + 16 * 72 + 0, 72);  s1 = WMMA_BF16(qf0, kf, s1);
      kf = load_frag_b(Ks + 16 * 72 + 32, 72); s1 = WMMA_BF16(qf1, kf, s1);
    }

    // Online softmax. Row r of this wave lives in lanes sharing (lane>>4).
    float p0[8], p1[8], mnew[8], alpha[8];
#pragma unroll
    for (int r = 0; r < 8; ++r) {
      float a = s0[r] * 0.125f;  // 1/sqrt(64)
      float c = s1[r] * 0.125f;
      if (causal) {
        int qrow = qbase + r + 8 * hi;
        if (kb * 32 + nn > qrow)      a = -3.0e38f;
        if (kb * 32 + 16 + nn > qrow) c = -3.0e38f;
      }
      p0[r] = a; p1[r] = c;
      mnew[r] = fmaxf(a, c);
    }
#pragma unroll
    for (int m = 1; m < 16; m <<= 1)
#pragma unroll
      for (int r = 0; r < 8; ++r)
        mnew[r] = fmaxf(mnew[r], __shfl_xor(mnew[r], m, 32));
#pragma unroll
    for (int r = 0; r < 8; ++r) {
      mnew[r]  = fmaxf(mnew[r], mrow[r]);
      alpha[r] = __expf(mrow[r] - mnew[r]);
      mrow[r]  = mnew[r];
      p0[r] = __expf(p0[r] - mnew[r]);
      p1[r] = __expf(p1[r] - mnew[r]);
    }
    float rs[8];
#pragma unroll
    for (int r = 0; r < 8; ++r) rs[r] = p0[r] + p1[r];
#pragma unroll
    for (int m = 1; m < 16; m <<= 1)
#pragma unroll
      for (int r = 0; r < 8; ++r) rs[r] += __shfl_xor(rs[r], m, 32);
#pragma unroll
    for (int r = 0; r < 8; ++r) {
      lrow[r] = lrow[r] * alpha[r] + rs[r];
      o0[r] *= alpha[r]; o1[r] *= alpha[r];
      o2[r] *= alpha[r]; o3[r] *= alpha[r];
      // re-layout P: C layout -> A layout via per-wave LDS strip
      Pw[(r + 8 * hi) * 48 + nn]      = (__bf16)p0[r];
      Pw[(r + 8 * hi) * 48 + 16 + nn] = (__bf16)p1[r];
    }

    // O += P(16x32) @ V(32x64): 4 output tiles over head dim
    v16bf pf = load_frag_a(Pw, 48);
    v16bf vf;
    vf = load_frag_b(Vt + 0 * 48, 48);  o0 = WMMA_BF16(pf, vf, o0);
    vf = load_frag_b(Vt + 16 * 48, 48); o1 = WMMA_BF16(pf, vf, o1);
    vf = load_frag_b(Vt + 32 * 48, 48); o2 = WMMA_BF16(pf, vf, o2);
    vf = load_frag_b(Vt + 48 * 48, 48); o3 = WMMA_BF16(pf, vf, o3);
  }

#pragma unroll
  for (int r = 0; r < 8; ++r) {
    float inv = 1.0f / lrow[r];
    size_t tokr = tok0 + r + 8 * hi;
    __bf16* op = O + tokr * 512 + (size_t)h * 64 + nn;
    op[0]  = (__bf16)(o0[r] * inv);
    op[16] = (__bf16)(o1[r] * inv);
    op[32] = (__bf16)(o2[r] * inv);
    op[48] = (__bf16)(o3[r] * inv);
  }
}

// ---------------------------------------------------------------------------
// LayerNorm over 512 dims; 1 wave per token; dual f32 + bf16 output.
// ---------------------------------------------------------------------------
__global__ __launch_bounds__(256) void k_ln(
    const float* __restrict__ X, const float* __restrict__ g,
    const float* __restrict__ bta, float* __restrict__ outF,
    __bf16* __restrict__ outB) {
  const int wid = threadIdx.x >> 5, lane = threadIdx.x & 31;
  const size_t tok = (size_t)blockIdx.x * 8 + wid;
  const float* xp = X + tok * 512;
  float xv[16], s = 0.0f, ss = 0.0f;
#pragma unroll
  for (int j = 0; j < 16; ++j) {
    float v = xp[j * 32 + lane];
    xv[j] = v; s += v; ss += v * v;
  }
#pragma unroll
  for (int m = 1; m < 32; m <<= 1) {
    s  += __shfl_xor(s, m, 32);
    ss += __shfl_xor(ss, m, 32);
  }
  float mu   = s * (1.0f / 512.0f);
  float var  = ss * (1.0f / 512.0f) - mu * mu;
  float rstd = rsqrtf(var + 1e-5f);
#pragma unroll
  for (int j = 0; j < 16; ++j) {
    int d = j * 32 + lane;
    float y = (xv[j] - mu) * rstd * g[d] + bta[d];
    outF[tok * 512 + d] = y;
    outB[tok * 512 + d] = (__bf16)y;
  }
}

// ---------------------------------------------------------------------------
// Host orchestration
// ---------------------------------------------------------------------------
extern "C" void kernel_launch(void* const* d_in, const int* in_sizes, int n_in,
                              void* d_out, int out_size, void* d_ws, size_t ws_size,
                              hipStream_t stream) {
  (void)in_sizes; (void)n_in; (void)out_size; (void)ws_size;
  constexpr int Bb = 16, L = 512, D = 512, FF = 2048, M = Bb * L;  // 8192 tokens

  const float* dec_x = (const float*)d_in[1];
  const float* dattn = (const float*)d_in[9];    // [6][8][512][512]
  const float* dlng  = (const float*)d_in[10];   // [6][3][512]
  const float* dlnb  = (const float*)d_in[11];
  const float* dffw1 = (const float*)d_in[12];   // [6][512][2048]
  const float* dffb1 = (const float*)d_in[13];   // [6][2048]
  const float* dffw2 = (const float*)d_in[14];   // [6][2048][512]
  const float* dffb2 = (const float*)d_in[15];   // [6][512]

  char* ws = (char*)d_ws;
  float*  xF   = (float*)(ws);                         // 16 MB activations f32
  float*  tmpF = (float*)(ws + (size_t)(16 << 20));    // 16 MB pre-LN sum
  __bf16* xB   = (__bf16*)(ws + (size_t)(32 << 20));   //  8 MB activations bf16
  __bf16* qB   = (__bf16*)(ws + (size_t)(40 << 20));   //  8 MB
  __bf16* kB   = (__bf16*)(ws + (size_t)(48 << 20));   //  8 MB
  __bf16* vB   = (__bf16*)(ws + (size_t)(56 << 20));   //  8 MB
  __bf16* cB   = (__bf16*)(ws + (size_t)(64 << 20));   //  8 MB attention ctx
  __bf16* hB   = (__bf16*)(ws + (size_t)(72 << 20));   // 32 MB FFN hidden

  {
    int total = M * D;
    k_add_pe<<<(total + 255) / 256, 256, 0, stream>>>(dec_x, xF, xB, total);
  }

  // GEMM flavors (template instantiations; epilogue is branch-free):
  auto gemm_plain = [&](const __bf16* A, const float* W, __bf16* oB, int n, int k) {
    dim3 grid(n / 128, M / 128);
    k_gemm<false, false, false, false, true><<<grid, 256, 0, stream>>>(
        A, W, nullptr, nullptr, nullptr, oB, M, n, k);
  };
  auto gemm_res = [&](const __bf16* A, const float* W, const float* res,
                      float* oF, int n, int k) {
    dim3 grid(n / 128, M / 128);
    k_gemm<false, false, true, true, false><<<grid, 256, 0, stream>>>(
        A, W, nullptr, res, oF, nullptr, M, n, k);
  };
  auto gemm_bias_relu = [&](const __bf16* A, const float* W, const float* bias,
                            __bf16* oB, int n, int k) {
    dim3 grid(n / 128, M / 128);
    k_gemm<true, true, false, false, true><<<grid, 256, 0, stream>>>(
        A, W, bias, nullptr, nullptr, oB, M, n, k);
  };
  auto gemm_bias_res = [&](const __bf16* A, const float* W, const float* bias,
                           const float* res, float* oF, int n, int k) {
    dim3 grid(n / 128, M / 128);
    k_gemm<true, false, true, true, false><<<grid, 256, 0, stream>>>(
        A, W, bias, res, oF, nullptr, M, n, k);
  };

  const dim3 agrid(L / 128, 8, Bb);  // (q-tiles, heads, batch)
  const size_t WSZ = (size_t)D * D;

  for (int i = 0; i < 6; ++i) {
    const float* AW = dattn + (size_t)i * 8 * WSZ;

    // --- causal self-attention ---
    gemm_plain(xB, AW + 0 * WSZ, qB, D, D);
    gemm_plain(xB, AW + 1 * WSZ, kB, D, D);
    gemm_plain(xB, AW + 2 * WSZ, vB, D, D);
    k_attn<<<agrid, 256, 0, stream>>>(qB, kB, vB, cB, 1);
    gemm_res(cB, AW + 3 * WSZ, xF, tmpF, D, D);
    k_ln<<<M / 8, 256, 0, stream>>>(tmpF, dlng + (size_t)(i * 3 + 0) * D,
                                    dlnb + (size_t)(i * 3 + 0) * D, xF, xB);

    // --- "cross" attention (reference uses kv=dec, non-causal) ---
    gemm_plain(xB, AW + 4 * WSZ, qB, D, D);
    gemm_plain(xB, AW + 5 * WSZ, kB, D, D);
    gemm_plain(xB, AW + 6 * WSZ, vB, D, D);
    k_attn<<<agrid, 256, 0, stream>>>(qB, kB, vB, cB, 0);
    gemm_res(cB, AW + 7 * WSZ, xF, tmpF, D, D);
    k_ln<<<M / 8, 256, 0, stream>>>(tmpF, dlng + (size_t)(i * 3 + 1) * D,
                                    dlnb + (size_t)(i * 3 + 1) * D, xF, xB);

    // --- FFN ---
    gemm_bias_relu(xB, dffw1 + (size_t)i * D * FF, dffb1 + (size_t)i * FF, hB, FF, D);
    gemm_bias_res(hB, dffw2 + (size_t)i * FF * D, dffb2 + (size_t)i * D,
                  xF, tmpF, D, FF);
    float* dstF = (i == 5) ? (float*)d_out : xF;
    k_ln<<<M / 8, 256, 0, stream>>>(tmpF, dlng + (size_t)(i * 3 + 2) * D,
                                    dlnb + (size_t)(i * 3 + 2) * D, dstF, xB);
  }
}
